// BasicCNN_LSTM_66434554134821
// MI455X (gfx1250) — matrix operand order repaired and checked by reading, not compiled, not run on hardware
//
#include <hip/hip_runtime.h>
#include <math.h>

// ---- CDNA5 WMMA fragment types (wave32) ----
typedef __attribute__((ext_vector_type(16))) __bf16 v16bf;
typedef __attribute__((ext_vector_type(8)))  float  v8f;

#define HW        112
#define ROWB      336           // 112*3 elements per image row
#define BANDROWS  28
#define STAGEROWS 30            // band + 1-row halo each side
#define STAGE_ELT (STAGEROWS*ROWB)   // 10080
#define KPAD      32            // 27 patch elements padded to one WMMA K-slab
#define NF        48
#define NBANDS    4
#define TSTEPS    24
#define BATCH     8
#define UNITS     32
#define DIN       80

// round-to-nearest-even f32 -> bf16 (bit twiddle; no cvt-instruction dependence)
static __device__ __forceinline__ unsigned short f2bf(float f) {
    union { float f; unsigned u; } c; c.f = f;
    unsigned r = c.u + 0x7FFFu + ((c.u >> 16) & 1u);
    return (unsigned short)(r >> 16);
}

// single-instruction ReLU: v_med3_f32(x, 0, +huge) == clamp(x, 0, inf),
// avoids LLVM's canonicalize+fmaxnum double-max on AMDGPU
static __device__ __forceinline__ float relu1(float x) {
    return __builtin_amdgcn_fmed3f(x, 0.0f, 3.0e38f);
}

// =====================================================================
// Kernel 1: fused Conv3x3(3->48) + bias + ReLU + partial avg-pool.
// One workgroup per (frame, band of 28 rows). WMMA bf16, f32 accum.
// =====================================================================
__global__ __launch_bounds__(256)
void conv_pool_wmma(const float* __restrict__ x,
                    const float* __restrict__ conv_w,
                    const float* __restrict__ conv_b,
                    float* __restrict__ pfeat)   // [192][4][48] band-partial sums
{
    __shared__ unsigned short sFrame[STAGE_ELT];  // band in bf16
    __shared__ unsigned short sW[KPAD * NF];      // weights, K padded to 32
    __shared__ float sRed[8 * 96];                // 8 waves x 3 tiles x 32 lanes

    const int frame = blockIdx.x;
    const int band  = blockIdx.y;
    const int tid   = threadIdx.x;
    const int lane  = tid & 31;                   // wave32
    const int wave  = tid >> 5;
    const float* __restrict__ xf = x + (size_t)frame * (HW * ROWB);

    // gfx1250 prefetch of this band's global region
    __builtin_prefetch(xf + (size_t)(band * BANDROWS) * ROWB + tid * 32, 0, 0);

    // ---- stage weights: K index = (ky*3+kx)*3 + c  == HWIO flat order ----
    for (int idx = tid; idx < KPAD * NF; idx += 256) {
        int k = idx / NF, f = idx - k * NF;
        sW[idx] = f2bf((k < 27) ? conv_w[k * NF + f] : 0.f);
    }
    // ---- stage band rows [band*28-1, band*28+29), zero halo outside frame ----
    for (int idx = tid; idx < STAGE_ELT; idx += 256) {
        int row = idx / ROWB;
        int col = idx - row * ROWB;
        int gy  = band * BANDROWS - 1 + row;
        float v = (gy >= 0 && gy < HW) ? xf[gy * ROWB + col] : 0.f;
        sFrame[idx] = f2bf(v);
    }
    __syncthreads();

    // ---- build B fragments (3 tiles of [32 x 16] bf16), contiguous K order, same as A:
    //      lane n (col), lanes<16 hold K 0..15, lanes>=16 hold K 16..31 ----
    union { unsigned short s[16]; v16bf v; } bu[3];
    const int n  = lane & 15;
    const int kb = (lane < 16) ? 0 : 16;
    #pragma unroll
    for (int t = 0; t < 3; ++t) {
        #pragma unroll
        for (int i = 0; i < 16; ++i)
            bu[t].s[i] = sW[(kb + i) * NF + t * 16 + n];
    }
    float cb[3];
    #pragma unroll
    for (int t = 0; t < 3; ++t) cb[t] = conv_b[t * 16 + n];

    float acc[3] = {0.f, 0.f, 0.f};
    const int m      = lane & 15;                 // A-row (pixel) for this lane
    const int kbaseA = (lane < 16) ? 0 : 16;      // A lane split matches B: K 0..15 / 16..31

    // ---- 196 tiles in this band (28 rows x 7 col-tiles), waves stride by 8 ----
    for (int tt = wave; tt < BANDROWS * 7; tt += 8) {
        int ly  = tt / 7;
        int x0  = (tt - ly * 7) * 16;
        int gx0 = x0 + m;

        // A fragment: 16 pixels x 32 patch elements (im2col gather from LDS)
        union { unsigned short s[16]; v16bf v; } au;
        #pragma unroll
        for (int i = 0; i < 16; ++i) {
            // lane<16: K -> 0..15 ; lane>=16: K -> 16..31
            int k = kbaseA + i;
            unsigned short v = 0;
            if (k < 27) {
                int ky = k / 9;
                int rm = k - ky * 9;
                int kx = rm / 3;
                int c  = rm - kx * 3;
                int gx = gx0 + kx - 1;
                if (gx >= 0 && gx < HW)
                    v = sFrame[(ly + ky) * ROWB + gx * 3 + c];
            }
            au.s[i] = v;
        }

        #pragma unroll
        for (int t = 0; t < 3; ++t) {
            v8f c;
            #pragma unroll
            for (int r = 0; r < 8; ++r) c[r] = cb[t];   // bias pre-loaded into C
            c = __builtin_amdgcn_wmma_f32_16x16x32_bf16(
                    false, au.v, false, bu[t].v, (short)0, c, false, false);
            // ReLU (single v_med3_f32 each) + pairwise pool reduction
            float p0 = relu1(c[0]) + relu1(c[1]);
            float p1 = relu1(c[2]) + relu1(c[3]);
            float p2 = relu1(c[4]) + relu1(c[5]);
            float p3 = relu1(c[6]) + relu1(c[7]);
            acc[t] += (p0 + p1) + (p2 + p3);
        }
    }

    // ---- deterministic reduction: lanes l and l+16 hold filter t*16+(l&15) ----
    #pragma unroll
    for (int t = 0; t < 3; ++t)
        sRed[wave * 96 + t * 32 + lane] = acc[t];
    __syncthreads();
    if (tid < NF) {
        int t = tid >> 4, nn = tid & 15;
        float s = 0.f;
        #pragma unroll
        for (int w = 0; w < 8; ++w)
            s += sRed[w * 96 + t * 32 + nn] + sRed[w * 96 + t * 32 + 16 + nn];
        pfeat[(frame * NBANDS + band) * NF + tid] = s;   // unnormalized band sum
    }
}

// =====================================================================
// Kernel 2: band reduction + 24-step recurrent scan (latency-bound).
// Single workgroup; thread (b,u) owns one state element. Keeps the
// reference's carry-order swap; skips the dead W4 branch.
// =====================================================================
__global__ __launch_bounds__(256)
void rnn_scan(const float* __restrict__ pfeat,
              const float* __restrict__ W1, const float* __restrict__ b1,
              const float* __restrict__ W2, const float* __restrict__ b2,
              const float* __restrict__ W3, const float* __restrict__ b3,
              float* __restrict__ out)
{
    __shared__ float sFeats[BATCH * TSTEPS * NF];   // 36 KB
    __shared__ float s0[BATCH * UNITS];             // read as cell_s
    __shared__ float s1[BATCH * UNITS];             // read as hidden_s
    __shared__ float z [BATCH * DIN];
    const int tid = threadIdx.x;

    for (int idx = tid; idx < BATCH * TSTEPS * NF; idx += 256) {
        int frame = idx / NF, f = idx - frame * NF;
        float s = 0.f;
        #pragma unroll
        for (int bnd = 0; bnd < NBANDS; ++bnd)
            s += pfeat[(frame * NBANDS + bnd) * NF + f];
        sFeats[idx] = s * (1.0f / 12544.0f);        // global average pool
    }
    s0[tid] = 0.f;
    s1[tid] = 0.f;
    __syncthreads();

    const int b = tid >> 5, u = tid & 31;
    float nh = 0.f;
    for (int t = 0; t < TSTEPS; ++t) {
        // z = concat(feats[:,t,:], hidden_s)
        for (int idx = tid; idx < BATCH * DIN; idx += 256) {
            int bb = idx / DIN, j = idx - bb * DIN;
            z[idx] = (j < NF) ? sFeats[(bb * TSTEPS + t) * NF + j]
                              : s1[bb * UNITS + (j - NF)];
        }
        __syncthreads();
        float d1 = b1[u], d2 = b2[u], d3 = b3[u];
        const float* __restrict__ zb = &z[b * DIN];
        #pragma unroll 4
        for (int k = 0; k < DIN; ++k) {
            float zv = zb[k];
            d1 = fmaf(zv, W1[k * UNITS + u], d1);
            d2 = fmaf(zv, W2[k * UNITS + u], d2);
            d3 = fmaf(zv, W3[k * UNITS + u], d3);
        }
        float cs = s0[b * UNITS + u];                     // "cell_s" (swap bug)
        float x1 = (1.f / (1.f + expf(-d1))) * cs;
        float x2 = 1.f / (1.f + expf(-d2));
        float x3 = tanhf(d3);
        float nc = x1 + x2 * x3;
        nh = nc * tanhf(nc);
        __syncthreads();
        s0[b * UNITS + u] = nh;                           // carry = (new_hidden,
        s1[b * UNITS + u] = nc;                           //          new_cell)
        __syncthreads();
    }
    out[b * UNITS + u] = nh;                              // outs[-1]
}

// =====================================================================
extern "C" void kernel_launch(void* const* d_in, const int* in_sizes, int n_in,
                              void* d_out, int out_size, void* d_ws, size_t ws_size,
                              hipStream_t stream) {
    const float* x      = (const float*)d_in[0];
    const float* conv_w = (const float*)d_in[1];
    const float* conv_b = (const float*)d_in[2];
    const float* W1     = (const float*)d_in[3];
    const float* b1     = (const float*)d_in[4];
    const float* W2     = (const float*)d_in[5];
    const float* b2     = (const float*)d_in[6];
    const float* W3     = (const float*)d_in[7];
    const float* b3     = (const float*)d_in[8];
    // d_in[9] (W4) / d_in[10] (b4): computed-but-unused in the reference; skipped.

    float* pfeat = (float*)d_ws;                 // 192*4*48 f32 = 144 KB scratch
    float* out   = (float*)d_out;                // [8,32] f32

    conv_pool_wmma<<<dim3(BATCH * TSTEPS, NBANDS), 256, 0, stream>>>(
        x, conv_w, conv_b, pfeat);
    rnn_scan<<<1, 256, 0, stream>>>(pfeat, W1, b1, W2, b2, W3, b3, out);

    (void)in_sizes; (void)n_in; (void)out_size; (void)ws_size;
}
